// InferenceLinear_35811437314721
// MI455X (gfx1250) — matrix operand order repaired and checked by reading, not compiled, not run on hardware
//
#include <hip/hip_runtime.h>

// Problem constants (match reference)
#define BATCH  64
#define NN     1024
#define KKDIM  1024
#define MMDIM  1024
#define NGROUP 16
#define GSIZE  64

// Tiling
#define MT 128      // m tile per block
#define NT 128      // n tile per block (8 waves x 16 rows)
#define KT 32       // k step (one WMMA depth)
#define LDS_STRIDE 40   // halves per m-row in LDS (32 data + 8 pad; 80B keeps 16B align)

typedef __attribute__((ext_vector_type(16))) __bf16 v16bf;
typedef __attribute__((ext_vector_type(2)))  __bf16 v2bf;
typedef __attribute__((ext_vector_type(8)))  float  v8f;
typedef __attribute__((ext_vector_type(4)))  float  f32x4;
typedef __attribute__((ext_vector_type(4)))  unsigned int u32x4;

union Frag16 {
  v16bf v;
  unsigned short h[16];
  unsigned int u[8];
  f32x4 q[2];
};

// Pack two floats into a bf16x2 dword (fuses to v_cvt_pk_bf16_f32)
__device__ __forceinline__ unsigned int pack2_bf16(float lo, float hi) {
  v2bf p;
  p.x = (__bf16)lo;
  p.y = (__bf16)hi;
  return __builtin_bit_cast(unsigned int, p);
}

// ---------------------------------------------------------------------------
// Kernel 1: dequantize W (uint8 affine, per-row 64-wide groups) -> bf16 once.
// 2 MB result is L2-resident (192 MB L2); amortized over 4096 consumer blocks.
// ---------------------------------------------------------------------------
__global__ __launch_bounds__(256)
void dequant_w_kernel(const unsigned char* __restrict__ qweight,
                      const float* __restrict__ qrange,
                      const float* __restrict__ qmin,
                      unsigned short* __restrict__ wbf) {
  const int idx  = blockIdx.x * 256 + threadIdx.x;  // one thread per 8 elements
  const int base = idx * 8;                          // flat index in N*K
  const int n = base >> 10;                          // / KKDIM
  const int k = base & (KKDIM - 1);
  const int g = k >> 6;                              // 8-chunk lies in one group
  const float scale = qrange[n * NGROUP + g];
  const float zero  = qmin[n * NGROUP + g];

  unsigned long long q = *(const unsigned long long*)(qweight + base);
  u32x4 o;
#pragma unroll
  for (int j = 0; j < 4; ++j) {
    float a = (float)((unsigned)(q >> (16 * j))     & 0xFFu) * scale + zero;
    float b = (float)((unsigned)(q >> (16 * j + 8)) & 0xFFu) * scale + zero;
    o[j] = pack2_bf16(a, b);
  }
  *(u32x4*)(wbf + base) = o;
}

// ---------------------------------------------------------------------------
// Kernel 2: C[b] = W @ X[b] via v_wmma_f32_16x16x32_bf16.
// PRE=true: A-fragments are raw b128 loads from pre-dequantized bf16 W (L2).
// PRE=false: fallback, dequantize inline (used only if workspace too small).
// ---------------------------------------------------------------------------
template <bool PRE>
__global__ __launch_bounds__(256)
void qlinear_wmma_bf16_kernel(const float* __restrict__ x,
                              const unsigned char* __restrict__ qweight,
                              const float* __restrict__ qrange,
                              const float* __restrict__ qmin,
                              const unsigned short* __restrict__ wbf,
                              float* __restrict__ out) {
  // X tile staged as bf16, [m][k] layout so B-fragments are 2x ds_load_b128/lane
  __shared__ __align__(16) unsigned short ldsx[MT * LDS_STRIDE];   // ~10 KB

  const int b    = blockIdx.z;
  const int m0   = blockIdx.x * MT;
  const int n0   = blockIdx.y * NT;
  const int tid  = threadIdx.x;
  const int wave = tid >> 5;        // wave32
  const int lane = tid & 31;
  const int l16  = lane & 15;
  const int lhalf = lane >> 4;      // 0: K 0..7 / 16..23, 1: K 8..15 / 24..31

  const float* xb = x + (size_t)b * KKDIM * MMDIM;

  // staging slots: 32(k) x 128(m) fp32 tile, 256 threads, 2 slots each
  const int kp0 = tid >> 5,         mq0 = tid & 31;
  const int kp1 = (tid + 256) >> 5, mq1 = (tid + 256) & 31;

  // A-side: this lane's weight row
  const int nrow = n0 + wave * 16 + l16;
  const unsigned char*  qrow = qweight + (size_t)nrow * KKDIM;
  const unsigned short* wrow = wbf + (size_t)nrow * KKDIM;

  v8f acc[8];
#pragma unroll
  for (int f = 0; f < 8; ++f) acc[f] = (v8f){0, 0, 0, 0, 0, 0, 0, 0};

  // prefetch first X tile into registers (double-buffer via regs)
  f32x4 sA0, sB0, sA1, sB1;
  {
    const float* p0 = xb + (size_t)(2 * kp0) * MMDIM + m0 + 4 * mq0;
    sA0 = *(const f32x4*)p0;
    sB0 = *(const f32x4*)(p0 + MMDIM);
    const float* p1 = xb + (size_t)(2 * kp1) * MMDIM + m0 + 4 * mq1;
    sA1 = *(const f32x4*)p1;
    sB1 = *(const f32x4*)(p1 + MMDIM);
  }

  for (int kk = 0; kk < KKDIM / KT; ++kk) {
    const int k0 = kk * KT;

    __syncthreads();   // previous tile fully consumed
    // commit staged tile: pack (k, k+1) bf16 pairs -> LDS [m][k]
#pragma unroll
    for (int i = 0; i < 4; ++i) {
      *(unsigned int*)&ldsx[(4 * mq0 + i) * LDS_STRIDE + 2 * kp0] =
          pack2_bf16(sA0[i], sB0[i]);
      *(unsigned int*)&ldsx[(4 * mq1 + i) * LDS_STRIDE + 2 * kp1] =
          pack2_bf16(sA1[i], sB1[i]);
    }
    __syncthreads();

    // issue next tile's global loads early (overlap with WMMA compute)
    if (kk + 1 < KKDIM / KT) {
      const float* p0 = xb + (size_t)(k0 + KT + 2 * kp0) * MMDIM + m0 + 4 * mq0;
      sA0 = *(const f32x4*)p0;
      sB0 = *(const f32x4*)(p0 + MMDIM);
      const float* p1 = xb + (size_t)(k0 + KT + 2 * kp1) * MMDIM + m0 + 4 * mq1;
      sA1 = *(const f32x4*)p1;
      sB1 = *(const f32x4*)(p1 + MMDIM);
    }

    // --- A fragment: 16 rows x 32 k of W in WMMA lane layout ---
    Frag16 afrag;
    const int ka = k0 + lhalf * 8;  // this half-wave's K octet base
    if constexpr (PRE) {
      // pre-dequantized bf16 weights: two global b128 loads, zero VALU (L2 hit)
      afrag.q[0] = *(const f32x4*)(wrow + ka);        // K ka..ka+7   -> VGPR 0..3
      afrag.q[1] = *(const f32x4*)(wrow + ka + 16);   // K ka+16..+23 -> VGPR 4..7
    } else {
      const int g = k0 >> 6;        // KT=32 lies inside one 64-wide quant group
      const float scale = qrange[nrow * NGROUP + g];
      const float zero  = qmin[nrow * NGROUP + g];
      unsigned long long q01 = *(const unsigned long long*)(qrow + ka);
      unsigned long long q23 = *(const unsigned long long*)(qrow + ka + 16);
#pragma unroll
      for (int j = 0; j < 4; ++j) {
        float w0a = (float)((unsigned)(q01 >> (16 * j))     & 0xFFu) * scale + zero;
        float w0b = (float)((unsigned)(q01 >> (16 * j + 8)) & 0xFFu) * scale + zero;
        float w1a = (float)((unsigned)(q23 >> (16 * j))     & 0xFFu) * scale + zero;
        float w1b = (float)((unsigned)(q23 >> (16 * j + 8)) & 0xFFu) * scale + zero;
        afrag.u[j]     = pack2_bf16(w0a, w0b);
        afrag.u[4 + j] = pack2_bf16(w1a, w1b);
      }
    }

    // --- 8 B fragments from LDS + 8 WMMAs, software double-buffered so the
    //     DS loads of fragment f+1 are in flight while WMMA f issues ---
    Frag16 bfrag[2];
    {
      const unsigned short* s0 = &ldsx[(l16)*LDS_STRIDE + lhalf * 8];
      bfrag[0].q[0] = *(const f32x4*)(s0);
      bfrag[0].q[1] = *(const f32x4*)(s0 + 16);
    }
#pragma unroll
    for (int f = 0; f < 8; ++f) {
      if (f + 1 < 8) {
        const unsigned short* sn =
            &ldsx[((f + 1) * 16 + l16) * LDS_STRIDE + lhalf * 8];
        bfrag[(f + 1) & 1].q[0] = *(const f32x4*)(sn);       // ds_load_b128
        bfrag[(f + 1) & 1].q[1] = *(const f32x4*)(sn + 16);  // ds_load_b128
      }
      acc[f] = __builtin_amdgcn_wmma_f32_16x16x32_bf16(
          false, afrag.v, false, bfrag[f & 1].v, (short)0, acc[f], false, false);
    }
  }

  // --- epilogue: C/D layout -> global (row = r + 8*lhalf, col = l16) ---
  float* ob = out + ((size_t)b * NN + n0 + wave * 16 + lhalf * 8) * MMDIM + m0 + l16;
#pragma unroll
  for (int f = 0; f < 8; ++f) {
#pragma unroll
    for (int r = 0; r < 8; ++r) {
      ob[(size_t)r * MMDIM + f * 16] = acc[f][r];
    }
  }
}

extern "C" void kernel_launch(void* const* d_in, const int* in_sizes, int n_in,
                              void* d_out, int out_size, void* d_ws, size_t ws_size,
                              hipStream_t stream) {
  const float*         x  = (const float*)d_in[0];
  const unsigned char* qw = (const unsigned char*)d_in[1];
  const float*         qr = (const float*)d_in[2];
  const float*         qm = (const float*)d_in[3];
  float*               out = (float*)d_out;

  dim3 grid(MMDIM / MT, NN / NT, BATCH);   // 8 x 8 x 64 blocks
  dim3 block(256);                          // 8 wave32 waves

  const size_t wbf_bytes = (size_t)NN * KKDIM * sizeof(unsigned short);  // 2 MB
  if (ws_size >= wbf_bytes) {
    unsigned short* wbf = (unsigned short*)d_ws;
    // one-shot dequant: N*K / (256 threads * 8 elem) = 512 blocks
    dequant_w_kernel<<<dim3((NN * KKDIM) / (256 * 8)), block, 0, stream>>>(
        qw, qr, qm, wbf);
    qlinear_wmma_bf16_kernel<true><<<grid, block, 0, stream>>>(
        x, qw, qr, qm, wbf, out);
  } else {
    qlinear_wmma_bf16_kernel<false><<<grid, block, 0, stream>>>(
        x, qw, qr, qm, (const unsigned short*)d_ws, out);
  }
}
